// Attention_12953621365472
// MI455X (gfx1250) — matrix operand order repaired and checked by reading, not compiled, not run on hardware
//
#include <hip/hip_runtime.h>
#include <hip/hip_bf16.h>

// Problem constants (match reference)
constexpr int Bc = 32;
constexpr int Nc = 2048;
constexpr int Dc = 512;
constexpr int Kc = 256;
constexpr float SLOPE = 0.01f;

typedef __attribute__((ext_vector_type(16))) __bf16 v16bf;
typedef __attribute__((ext_vector_type(8)))  float  v8f;

union FragAB {
    unsigned int u[8];
    v16bf v;
};

// -------------------------------------------------------------------------
// Kernel 1: vQp[b,k] = vQ[b,:] . Wq[:,k] + bq[k]   (tiny: 32x256)
// -------------------------------------------------------------------------
__global__ __launch_bounds__(256)
void vqp_kernel(const float* __restrict__ vQ, const float* __restrict__ Wq,
                const float* __restrict__ bq, float* __restrict__ vQp) {
    const int b = blockIdx.x;
    const int k = threadIdx.x;
    float acc = bq[k];
    const float* q = vQ + (size_t)b * Dc;
#pragma unroll 4
    for (int d = 0; d < Dc; ++d)
        acc = fmaf(q[d], Wq[(size_t)d * Kc + k], acc);
    vQp[b * Kc + k] = acc;
}

// -------------------------------------------------------------------------
// Kernel 2: per (b, 64-row tile):
//   vIp = vI @ Wi (bf16 WMMA, f32 accumulate)  -> stored to workspace
//   scores[n] = sum_k leakyrelu(vIp + vQp) * Wp[k] + bp  (fused, from regs)
// Block: 256 threads (8 waves). Wave w: m_tile = w>>1 (16 rows),
// n_half = w&1 (128 cols) -> 8 WMMA C tiles per wave.
// -------------------------------------------------------------------------
__global__ __launch_bounds__(256)
void gemm_score_kernel(const float* __restrict__ vI, const float* __restrict__ Wi,
                       const float* __restrict__ vQp, const float* __restrict__ Wp,
                       const float* __restrict__ bp,
                       float* __restrict__ vIp_ws, float* __restrict__ scores_ws) {
    // stride 36 halves (72 B): 16 consecutive rows/cols land in 16 distinct
    // LDS banks -> conflict-free 32-bit fragment loads
    __shared__ __bf16 sA[64 * 36];     // 64 rows x 32 D (bf16)
    __shared__ __bf16 sWiT[256 * 36];  // 256 cols x 32 D (bf16, transposed)
    __shared__ float  sScore[64];

    const int tid  = threadIdx.x;
    const int lane = tid & 31;
    const int wave = tid >> 5;
    const int b    = blockIdx.y;
    const int n0   = blockIdx.x * 64;

    const int m_tile = wave >> 1;                 // 0..3
    const int n_half = wave & 1;                  // 0..1
    const int row    = lane & 15;
    const int ksbaseA = (lane < 16) ? 0 : 16;     // A K-half select
    const int kbaseB  = (lane < 16) ? 0 : 16;     // B K-half select

    // ---- staging addressing, hoisted out of the k-loop ----
    // vI tile: thread (r0 = tid>>5, c = tid&31) loads rows r0, r0+8, ..., r0+56
    const int rA = tid >> 5;
    const int cA = tid & 31;
    const float* gA = vI + (size_t)(b * Nc + n0 + rA) * Dc + cA;   // +32 per step
    __bf16*      lA = sA + rA * 36 + cA;
    // Wi chunk: thread owns column tid, loads d-rows j = 0..31
    const float* gW = Wi + tid;                                    // +32*Kc per step
    __bf16*      lW = sWiT + tid * 36;

    v8f acc[8];
    const v8f vzero = {0.f, 0.f, 0.f, 0.f, 0.f, 0.f, 0.f, 0.f};
#pragma unroll
    for (int t = 0; t < 8; ++t) acc[t] = vzero;

    const unsigned int* sA32 = (const unsigned int*)sA;
    const unsigned int* sW32 = (const unsigned int*)sWiT;
    const int abase = (m_tile * 16 + row) * 18 + (ksbaseA >> 1);

    for (int kt = 0; kt < Dc / 32; ++kt) {
        // register-stage global data (overlaps previous iteration's WMMAs)
        float va[8], vw[32];
#pragma unroll
        for (int j = 0; j < 8; ++j)  va[j] = gA[(size_t)j * 8 * Dc];
#pragma unroll
        for (int j = 0; j < 32; ++j) vw[j] = gW[(size_t)j * Kc];
        if (kt + 2 < Dc / 32) {
            __builtin_prefetch(gA + 64, 0, 0);              // global_prefetch_b8
            __builtin_prefetch(gW + (size_t)64 * Kc, 0, 0);
        }
        gA += 32;
        gW += (size_t)32 * Kc;

        __syncthreads();   // previous iteration done reading LDS
#pragma unroll
        for (int j = 0; j < 8; ++j)  lA[j * 8 * 36] = (__bf16)va[j];
#pragma unroll
        for (int j = 0; j < 32; ++j) lW[j] = (__bf16)vw[j];
        __syncthreads();   // LDS tile ready

        // A fragment: 16x32 bf16. Lane holds row; VGPR0-7: K=ks..ks+15
        // (ks = 0 for lanes 0-15, 16 for 16-31).
        FragAB a;
#pragma unroll
        for (int p = 0; p < 4; ++p) a.u[p]     = sA32[abase + p];
#pragma unroll
        for (int p = 0; p < 4; ++p) a.u[4 + p] = sA32[abase + 4 + p];

        // hoist ALL B fragments (one DS clause), then back-to-back WMMAs
        FragAB bb[8];
#pragma unroll
        for (int t = 0; t < 8; ++t) {
            const int bbase = (n_half * 128 + t * 16 + row) * 18 + (kbaseB >> 1);
#pragma unroll
            for (int p = 0; p < 8; ++p) bb[t].u[p] = sW32[bbase + p];
        }
#pragma unroll
        for (int t = 0; t < 8; ++t)
            acc[t] = __builtin_amdgcn_wmma_f32_16x16x32_bf16(
                false, a.v, false, bb[t].v, (short)0, acc[t], false, false);
    }

    // ---- fused epilogue: spill vIp, reduce leakyrelu(vIp+vQp).Wp into scores
    if (tid < 64) sScore[tid] = 0.f;
    __syncthreads();

    const int mrow_off = (lane < 16) ? 0 : 8;  // C/D layout: VGPR v -> M=v or v+8
    float rsum[8];
#pragma unroll
    for (int v = 0; v < 8; ++v) rsum[v] = 0.f;

#pragma unroll
    for (int t = 0; t < 8; ++t) {
        const int col = n_half * 128 + t * 16 + row;
        const float wp = Wp[col];
        const float q  = vQp[b * Kc + col];
#pragma unroll
        for (int v = 0; v < 8; ++v) {
            const float val = acc[t][v];
            const int m = m_tile * 16 + mrow_off + v;
            vIp_ws[(size_t)(b * Nc + n0 + m) * Kc + col] = val;
            float h = val + q;
            h = (h > 0.f) ? h : SLOPE * h;
            rsum[v] = fmaf(h, wp, rsum[v]);
        }
    }
#pragma unroll
    for (int v = 0; v < 8; ++v)
        atomicAdd(&sScore[m_tile * 16 + mrow_off + v], rsum[v]);
    __syncthreads();

    if (tid < 64)
        scores_ws[b * Nc + n0 + tid] = sScore[tid] + bp[0];
}

// -------------------------------------------------------------------------
// Kernel 3: per batch b — softmax over N, then out[b,k] =
//   (sum_n pi[n] * vIp[b,n,k]) + vQp[b,k]
// 4 groups of 64 lanes; each lane owns 4 consecutive k -> float4 (b128) loads
// -------------------------------------------------------------------------
__global__ __launch_bounds__(256)
void softmax_attend_kernel(const float* __restrict__ scores,
                           const float* __restrict__ vIp,
                           const float* __restrict__ vQp,
                           float* __restrict__ out) {
    __shared__ float sS[Nc];
    __shared__ float sRed[256];
    __shared__ float sPart[4 * 256];
    const int b = blockIdx.x;
    const int tid = threadIdx.x;

    for (int i = tid; i < Nc; i += 256) sS[i] = scores[b * Nc + i];
    __syncthreads();

    float mx = -3.4e38f;
    for (int i = tid; i < Nc; i += 256) mx = fmaxf(mx, sS[i]);
    sRed[tid] = mx;
    __syncthreads();
    for (int off = 128; off > 0; off >>= 1) {
        if (tid < off) sRed[tid] = fmaxf(sRed[tid], sRed[tid + off]);
        __syncthreads();
    }
    mx = sRed[0];
    __syncthreads();

    float ls = 0.f;
    for (int i = tid; i < Nc; i += 256) {
        const float e = __expf(sS[i] - mx);
        sS[i] = e;
        ls += e;
    }
    sRed[tid] = ls;
    __syncthreads();
    for (int off = 128; off > 0; off >>= 1) {
        if (tid < off) sRed[tid] += sRed[tid + off];
        __syncthreads();
    }
    const float inv = 1.f / sRed[0];

    // weighted sum over N, float4-per-lane, split N into 4 chunks
    const int g = tid >> 6;            // n-chunk 0..3 (each Nc/4 = 512 rows)
    const int l = tid & 63;            // owns k = 4l .. 4l+3
    const float4* base =
        (const float4*)(vIp + (size_t)b * Nc * Kc + (size_t)g * (Nc / 4) * Kc) + l;
    const float* w = sS + g * (Nc / 4);
    float4 av = {0.f, 0.f, 0.f, 0.f};
#pragma unroll 4
    for (int n = 0; n < Nc / 4; ++n) {
        const float  p = w[n];
        const float4 x = base[(size_t)n * (Kc / 4)];
        av.x = fmaf(p, x.x, av.x);
        av.y = fmaf(p, x.y, av.y);
        av.z = fmaf(p, x.z, av.z);
        av.w = fmaf(p, x.w, av.w);
    }
    sPart[g * 256 + l * 4 + 0] = av.x;
    sPart[g * 256 + l * 4 + 1] = av.y;
    sPart[g * 256 + l * 4 + 2] = av.z;
    sPart[g * 256 + l * 4 + 3] = av.w;
    __syncthreads();

    const float r = sPart[tid] + sPart[256 + tid] + sPart[512 + tid] + sPart[768 + tid];
    out[b * Kc + tid] = r * inv + vQp[b * Kc + tid];
}

// -------------------------------------------------------------------------
extern "C" void kernel_launch(void* const* d_in, const int* in_sizes, int n_in,
                              void* d_out, int out_size, void* d_ws, size_t ws_size,
                              hipStream_t stream) {
    (void)in_sizes; (void)n_in; (void)out_size; (void)ws_size;
    const float* vI = (const float*)d_in[0];   // [B,N,D]
    const float* vQ = (const float*)d_in[1];   // [B,D]
    const float* Wi = (const float*)d_in[2];   // [D,K]
    const float* Wq = (const float*)d_in[3];   // [D,K]
    const float* bq = (const float*)d_in[4];   // [K]
    const float* Wp = (const float*)d_in[5];   // [K,1]
    const float* bp = (const float*)d_in[6];   // [1]
    float* out = (float*)d_out;                // [B,K]

    float* ws     = (float*)d_ws;
    float* vQp    = ws;                         // B*K
    float* scores = ws + Bc * Kc;               // B*N
    float* vIp    = ws + Bc * Kc + Bc * Nc;     // B*N*K (64 MiB)

    vqp_kernel<<<Bc, 256, 0, stream>>>(vQ, Wq, bq, vQp);
    gemm_score_kernel<<<dim3(Nc / 64, Bc), 256, 0, stream>>>(
        vI, Wi, vQp, Wp, bp, vIp, scores);
    softmax_attend_kernel<<<Bc, 256, 0, stream>>>(scores, vIp, vQp, out);
}
